// MultiHeadAttention_82884278878380
// MI455X (gfx1250) — matrix operand order repaired and checked by reading, not compiled, not run on hardware
//
#include <hip/hip_runtime.h>

// Problem constants (match reference)
#define B_  4
#define S_  2048
#define D_  1024
#define H_  16
#define DK_ 64

typedef __attribute__((ext_vector_type(16))) _Float16 v16h;
typedef __attribute__((ext_vector_type(8)))  _Float16 v8h;
typedef __attribute__((ext_vector_type(4)))  _Float16 v4h;
typedef __attribute__((ext_vector_type(8)))  float    v8f;
typedef __attribute__((ext_vector_type(4)))  int      v4i;
typedef __attribute__((ext_vector_type(8)))  int      v8i;
typedef __attribute__((ext_vector_type(4)))  unsigned int v4u;

// ---------------------------------------------------------------------------
// CDNA5 async global->LDS copy (ASYNCcnt-tracked, bypasses VGPRs).
// ---------------------------------------------------------------------------
#if defined(__HIP_DEVICE_COMPILE__) && \
    __has_builtin(__builtin_amdgcn_global_load_async_to_lds_b128) && \
    __has_builtin(__builtin_amdgcn_s_wait_asynccnt)
#define HAS_ASYNC_LDS 1
#else
#define HAS_ASYNC_LDS 0
#endif

// CDNA5 Tensor Data Mover (TENSORcnt-tracked DMA, ISA ch. 7.2 / 8).
#if defined(__HIP_DEVICE_COMPILE__) && \
    __has_builtin(__builtin_amdgcn_tensor_load_to_lds) && \
    __has_builtin(__builtin_amdgcn_s_wait_tensorcnt)
#define HAS_TDM 1
#else
#define HAS_TDM 0
#endif

static __device__ inline void async_copy_b128(const _Float16* g, _Float16* l) {
#if HAS_ASYNC_LDS
  __builtin_amdgcn_global_load_async_to_lds_b128(
      (__attribute__((address_space(1))) v4i*)g,
      (__attribute__((address_space(3))) v4i*)l, 0, 0);
#else
  *(v8h*)l = *(const v8h*)g;
#endif
}

static __device__ inline void async_wait_all() {
#if HAS_ASYNC_LDS
  __builtin_amdgcn_s_wait_asynccnt(0);
#endif
}

// ---------------------------------------------------------------------------
// TDM tile load: ROWS x 32 halves from row-major source (row stride =
// rowStrideHalves) into LDS at byte offset lds_off, with HW padding of
// 16 B after every 64 B row => effective LDS row stride 80 B (= 40 halves),
// matching the WMMA fragment loaders below.
// D# layout per CDNA5 ISA 8.3 (group 0) / 8.4 (group 1).
// This toolchain's builtin takes 6 args (g0, g1, g2, g3, g4, cpol).
// ---------------------------------------------------------------------------
static __device__ inline void tdm_load_tile_k32(const _Float16* gsrc,
                                                unsigned lds_off, int rows,
                                                int rowStrideHalves) {
#if HAS_TDM
  const unsigned long long ga = (unsigned long long)(size_t)gsrc;
  v4u g0;
  g0[0] = 1u;                                   // count=1, user descriptor
  g0[1] = lds_off;                              // lds_addr (bytes)
  g0[2] = (unsigned)(ga & 0xFFFFFFFFu);         // global_addr[31:0]
  g0[3] = (unsigned)((ga >> 32) & 0x01FFFFFFu)  // global_addr[56:32]
          | (2u << 30);                         // type=2 ("image")
  v8i g1;
  g1[0] = (1 << 16)                             // data_size: 2 bytes
        | (1 << 20)                             // pad_enable
        | (3 << 22)                             // pad_interval: 16 DWORDs (64 B)
        | (3 << 25);                            // pad_amount: 4 DWORDs (16 B)
  g1[1] = (rowStrideHalves & 0xFFFF) << 16;                       // tensor_dim0[15:0]
  g1[2] = ((rowStrideHalves >> 16) & 0xFFFF) | ((rows & 0xFFFF) << 16); // dim0 hi | dim1 lo
  g1[3] = (32 << 16);                           // tensor_dim1 hi=0 | tile_dim0=32
  g1[4] = rows & 0xFFFF;                        // tile_dim1 | tile_dim2=0
  g1[5] = rowStrideHalves;                      // tensor_dim0_stride[31:0]
  g1[6] = 0;                                    // stride hi | dim1_stride lo
  g1[7] = 0;
  const v4i z4 = {0, 0, 0, 0};                  // groups 2/3 unused (2D tensor)
  const v8i z8 = {0, 0, 0, 0, 0, 0, 0, 0};
  __builtin_amdgcn_tensor_load_to_lds(g0, g1, z4, z4, z8, 0);
#endif
}

static __device__ inline void tdm_wait_all() {
#if HAS_TDM
  __builtin_amdgcn_s_wait_tensorcnt(0);
#endif
}

// ---------------------------------------------------------------------------
// WMMA helpers: layouts per CDNA5 ISA 7.12.2 (wave32)
//  A (16x32 f16): lane holds row M=lane&15; K chunks {kb..kb+7, 16+kb..16+kb+7}
//                 with kb = 8*(lane>>4)
//  B (32x16 f16): lane holds col N=lane&15; K chunks same as A (kb, 16+kb)
//  C/D (16x16 f32): lane holds col N=lane&15; rows M = e + 8*(lane>>4)
// ---------------------------------------------------------------------------
static __device__ inline v16h make_v16(v8h lo, v8h hi) {
  v16h a;
#pragma unroll
  for (int i = 0; i < 8; ++i) { a[i] = lo[i]; a[i + 8] = hi[i]; }
  return a;
}

static __device__ inline v16h lds_load_a16(const _Float16* p, int stride,
                                           int row0, int kofs, int lane) {
  const int r  = row0 + (lane & 15);
  const int kb = (lane >> 4) * 8;
  const _Float16* base = p + r * stride + kofs;
  v8h lo = *(const v8h*)(base + kb);
  v8h hi = *(const v8h*)(base + 16 + kb);
  return make_v16(lo, hi);
}

static __device__ inline v16h lds_load_b16(const _Float16* p, int stride,
                                           int col0, int kofs, int lane) {
  const int c  = col0 + (lane & 15);
  const int kb = (lane >> 4) * 8;
  const _Float16* base = p + c * stride + kofs;
  v8h lo = *(const v8h*)(base + kb);
  v8h hi = *(const v8h*)(base + 16 + kb);
  return make_v16(lo, hi);
}

static __device__ inline v8f wmma_f32_f16(v16h a, v16h b, v8f c) {
  return __builtin_amdgcn_wmma_f32_16x16x32_f16(false, a, false, b,
                                                (short)0, c, false, false);
}

// ---------------------------------------------------------------------------
// Kernel 1a: f32 -> f16 convert (4 elems/thread)
// ---------------------------------------------------------------------------
__global__ __launch_bounds__(256) void cvt_f16_kernel(
    const float* __restrict__ in, _Float16* __restrict__ out, int n4) {
  int i = blockIdx.x * 256 + threadIdx.x;
  if (i < n4) {
    float4 v = ((const float4*)in)[i];
    v4h o;
    o[0] = (_Float16)v.x; o[1] = (_Float16)v.y;
    o[2] = (_Float16)v.z; o[3] = (_Float16)v.w;
    ((v4h*)out)[i] = o;
  }
}

// ---------------------------------------------------------------------------
// Kernel 1b: W [K][N] f32 -> WT [N][K] f16 (tiled 32x32 transpose)
// ---------------------------------------------------------------------------
__global__ __launch_bounds__(256) void cvt_wT_kernel(
    const float* __restrict__ W, _Float16* __restrict__ WT) {
  __shared__ float tile[32][33];
  const int k0 = blockIdx.x * 32, n0 = blockIdx.y * 32;
  const int tx = threadIdx.x & 31, ty = threadIdx.x >> 5;  // ty 0..7
#pragma unroll
  for (int i = ty; i < 32; i += 8)
    tile[i][tx] = W[(size_t)(k0 + i) * D_ + n0 + tx];
  __syncthreads();
#pragma unroll
  for (int i = ty; i < 32; i += 8)
    WT[(size_t)(n0 + i) * D_ + k0 + tx] = (_Float16)tile[tx][i];
}

// ---------------------------------------------------------------------------
// Kernel 2/4: GEMM  out[M,N] = A[M,K](f16) @ BT[N,K](f16)^T + bias
// Block tile 128x128, K-step 32. TDM (tensor_load_to_lds) double-buffered
// staging issued by wave 0; fallback = per-lane async global->LDS.
// 8 waves; wave (wm,wn) owns a 64x32 region = 4(M) x 2(N) WMMA tiles.
// mode 0..2: write f16 permuted to [B,H,S,DK]; mode 3: write f32 flat [M,N].
// ---------------------------------------------------------------------------
__global__ __launch_bounds__(256) void gemm_f16_kernel(
    const _Float16* __restrict__ A, const _Float16* __restrict__ BT,
    const float* __restrict__ bias, _Float16* __restrict__ outPerm,
    float* __restrict__ outFlat, int M, int N, int K, int mode) {
  __shared__ __align__(16) _Float16 lA[2][128 * 40];
  __shared__ __align__(16) _Float16 lB[2][128 * 40];

  const int t    = threadIdx.x;
  const int lane = t & 31;
  const int wave = t >> 5;
  const int wm   = wave >> 2;  // 0..1 -> 64 rows each
  const int wn   = wave & 3;   // 0..3 -> 32 cols each
  const int m0   = blockIdx.x * 128;
  const int n0   = blockIdx.y * 128;

  v8f acc[4][2];
#pragma unroll
  for (int i = 0; i < 4; ++i)
#pragma unroll
    for (int j = 0; j < 2; ++j)
#pragma unroll
      for (int e = 0; e < 8; ++e) acc[i][j][e] = 0.0f;

  const int sr = t >> 2;         // 0..63
  const int sc = (t & 3) * 8;    // 0,8,16,24 halves

  auto stage = [&](int buf, int k0) {
#if HAS_TDM
    if (wave == 0) {  // one DMA descriptor per tile, issued by one wave
      tdm_load_tile_k32(&A [(size_t)m0 * K + k0],
                        (unsigned)(size_t)(void*)&lA[buf][0], 128, K);
      tdm_load_tile_k32(&BT[(size_t)n0 * K + k0],
                        (unsigned)(size_t)(void*)&lB[buf][0], 128, K);
    }
#else
    async_copy_b128(&A [(size_t)(m0 + sr)      * K + k0 + sc], &lA[buf][sr * 40 + sc]);
    async_copy_b128(&A [(size_t)(m0 + sr + 64) * K + k0 + sc], &lA[buf][(sr + 64) * 40 + sc]);
    async_copy_b128(&BT[(size_t)(n0 + sr)      * K + k0 + sc], &lB[buf][sr * 40 + sc]);
    async_copy_b128(&BT[(size_t)(n0 + sr + 64) * K + k0 + sc], &lB[buf][(sr + 64) * 40 + sc]);
#endif
  };
  auto stage_wait = [&]() {
#if HAS_TDM
    tdm_wait_all();     // no-op for waves with TENSORcnt == 0
#else
    async_wait_all();
#endif
  };

  stage(0, 0);
  stage_wait();
  __syncthreads();

  int cur = 0;
  for (int k0 = 0; k0 < K; k0 += 32) {
    if (k0 + 32 < K) stage(cur ^ 1, k0 + 32);  // overlap next fetch w/ compute

    v16h bfrag[2];
#pragma unroll
    for (int j = 0; j < 2; ++j)
      bfrag[j] = lds_load_b16(lB[cur], 40, wn * 32 + j * 16, 0, lane);
#pragma unroll
    for (int i = 0; i < 4; ++i) {
      v16h afrag = lds_load_a16(lA[cur], 40, wm * 64 + i * 16, 0, lane);
#pragma unroll
      for (int j = 0; j < 2; ++j)
        acc[i][j] = wmma_f32_f16(afrag, bfrag[j], acc[i][j]);
    }

    stage_wait();       // next-tile DMA landed (issuing wave)
    __syncthreads();    // everyone done reading cur; next tile visible to all
    cur ^= 1;
  }

  // Epilogue
  const int rr = lane >> 4;
  const int cc = lane & 15;
#pragma unroll
  for (int i = 0; i < 4; ++i) {
#pragma unroll
    for (int j = 0; j < 2; ++j) {
      const int ncol = n0 + wn * 32 + j * 16 + cc;
      const float bv = bias[ncol];
#pragma unroll
      for (int e = 0; e < 8; ++e) {
        const int mrow = m0 + wm * 64 + i * 16 + e + 8 * rr;
        const float val = acc[i][j][e] + bv;
        if (mode < 3) {
          // token mrow = b*S + s ; col ncol = h*DK + dk -> [B,H,S,DK] f16
          const int b = mrow >> 11, s = mrow & (S_ - 1);
          const int h = ncol >> 6,  dk = ncol & (DK_ - 1);
          outPerm[(((size_t)(b * H_ + h) * S_ + s) * DK_) + dk] = (_Float16)val;
        } else {
          outFlat[(size_t)mrow * N + ncol] = val;
        }
      }
    }
  }
}

// ---------------------------------------------------------------------------
// Kernel 3: flash attention. Block = 128 query rows of one (b,h); 8 waves,
// each wave owns 16 rows end-to-end (online softmax stats stay in-wave).
// Q/K tiles staged with async global->LDS; V transposed via registers.
// ---------------------------------------------------------------------------
__global__ __launch_bounds__(256) void flash_attn_kernel(
    const _Float16* __restrict__ Q, const _Float16* __restrict__ Km,
    const _Float16* __restrict__ V, const int* __restrict__ mask,
    _Float16* __restrict__ O, float scale) {
  __shared__ __align__(16) _Float16 lQ[128 * 72];
  __shared__ __align__(16) _Float16 lK[64 * 72];
  __shared__ __align__(16) _Float16 lVt[64 * 72];   // [d][kk]
  __shared__ __align__(16) _Float16 lP[128 * 72];   // wave-private 16-row slabs

  const int t = threadIdx.x, lane = t & 31, wave = t >> 5;
  const int bh = blockIdx.y;
  const int b = bh >> 4, h = bh & 15;
  const int q0 = blockIdx.x * 128;

  const _Float16* Qg = Q + ((size_t)bh * S_ + q0) * DK_;
  const _Float16* Kg = Km + (size_t)bh * S_ * DK_;
  const _Float16* Vg = V + (size_t)bh * S_ * DK_;
  const int* maskB = mask + (size_t)b * S_ * S_;

  // Stage Q tile: 128 rows x 64 halves (async; waited in first loop iter)
#pragma unroll
  for (int i = 0; i < 4; ++i) {
    const int idx = t + i * 256;               // 0..1023
    const int r = idx >> 3, seg = (idx & 7) * 8;
    async_copy_b128(&Qg[(size_t)r * DK_ + seg], &lQ[r * 72 + seg]);
  }

  const int rr = lane >> 4, cc = lane & 15;
  v8f accO[4];
  float m_run[8], l_run[8];
#pragma unroll
  for (int dt = 0; dt < 4; ++dt)
#pragma unroll
    for (int e = 0; e < 8; ++e) accO[dt][e] = 0.0f;
#pragma unroll
  for (int e = 0; e < 8; ++e) { m_run[e] = -1e30f; l_run[e] = 0.0f; }

  _Float16* myP = lP + wave * 16 * 72;

  for (int kb0 = 0; kb0 < S_; kb0 += 64) {
    __syncthreads();  // all waves done reading previous K/Vt tiles
    // Stage K tile 64x64 ([kk][d] == B^T layout for Q@K^T) -- async
#pragma unroll
    for (int i = 0; i < 2; ++i) {
      const int idx = t + i * 256;
      const int r = idx >> 3, seg = (idx & 7) * 8;
      async_copy_b128(&Kg[(size_t)(kb0 + r) * DK_ + seg], &lK[r * 72 + seg]);
    }
    // Stage V transposed: lVt[d][kk] (vector global read, scalar LDS scatter)
#pragma unroll
    for (int i = 0; i < 2; ++i) {
      const int idx = t + i * 256;             // 0..511
      const int kk = idx >> 3, dseg = (idx & 7) * 8;
      v8h vv = *(const v8h*)&Vg[(size_t)(kb0 + kk) * DK_ + dseg];
#pragma unroll
      for (int j = 0; j < 8; ++j) lVt[(dseg + j) * 72 + kk] = vv[j];
    }
    async_wait_all();
    __syncthreads();

    // S = Q @ K^T  (wave's 16 rows x 64 cols)
    v8f sfrag[4];
#pragma unroll
    for (int nt = 0; nt < 4; ++nt)
#pragma unroll
      for (int e = 0; e < 8; ++e) sfrag[nt][e] = 0.0f;
#pragma unroll
    for (int ks = 0; ks < 64; ks += 32) {
      v16h aq = lds_load_a16(lQ, 72, wave * 16, ks, lane);
#pragma unroll
      for (int nt = 0; nt < 4; ++nt) {
        v16h bk = lds_load_b16(lK, 72, nt * 16, ks, lane);
        sfrag[nt] = wmma_f32_f16(aq, bk, sfrag[nt]);
      }
    }

    // Mask + scale, row max
    float sv[4][8], mnew[8];
#pragma unroll
    for (int e = 0; e < 8; ++e) mnew[e] = -1e30f;
#pragma unroll
    for (int nt = 0; nt < 4; ++nt) {
#pragma unroll
      for (int e = 0; e < 8; ++e) {
        const int qg = q0 + wave * 16 + e + 8 * rr;
        const int kg = kb0 + nt * 16 + cc;
        const int mv = maskB[(size_t)qg * S_ + kg];
        const float x = (mv != 0) ? sfrag[nt][e] * scale : -1e9f;
        sv[nt][e] = x;
        mnew[e] = fmaxf(mnew[e], x);
      }
    }
#pragma unroll
    for (int off = 1; off < 16; off <<= 1)
#pragma unroll
      for (int e = 0; e < 8; ++e)
        mnew[e] = fmaxf(mnew[e], __shfl_xor(mnew[e], off, 32));

    float fac[8], lsum[8];
#pragma unroll
    for (int e = 0; e < 8; ++e) {
      const float m2 = fmaxf(m_run[e], mnew[e]);
      fac[e] = __expf(m_run[e] - m2);
      m_run[e] = m2;
      lsum[e] = 0.0f;
    }

    // P = exp(S - m): write wave-private LDS slab in A-layout [q][kk]
#pragma unroll
    for (int nt = 0; nt < 4; ++nt) {
#pragma unroll
      for (int e = 0; e < 8; ++e) {
        const float p = __expf(sv[nt][e] - m_run[e]);
        lsum[e] += p;
        myP[(e + 8 * rr) * 72 + nt * 16 + cc] = (_Float16)p;
      }
    }
#pragma unroll
    for (int off = 1; off < 16; off <<= 1)
#pragma unroll
      for (int e = 0; e < 8; ++e) lsum[e] += __shfl_xor(lsum[e], off, 32);
#pragma unroll
    for (int e = 0; e < 8; ++e) l_run[e] = l_run[e] * fac[e] + lsum[e];
#pragma unroll
    for (int dt = 0; dt < 4; ++dt)
#pragma unroll
      for (int e = 0; e < 8; ++e) accO[dt][e] *= fac[e];

    // O += P @ V  (K-dim = 64 keys, two 32-chunks)
#pragma unroll
    for (int ks = 0; ks < 64; ks += 32) {
      v16h ap = lds_load_a16(myP, 72, 0, ks, lane);
#pragma unroll
      for (int dt = 0; dt < 4; ++dt) {
        v16h bv = lds_load_b16(lVt, 72, dt * 16, ks, lane);
        accO[dt] = wmma_f32_f16(ap, bv, accO[dt]);
      }
    }
  }

  // Normalize + store O as f16 in [B,S,D] (d = h*DK + dk)
#pragma unroll
  for (int e = 0; e < 8; ++e) {
    const float inv = 1.0f / l_run[e];
    const int qg = q0 + wave * 16 + e + 8 * rr;
#pragma unroll
    for (int dt = 0; dt < 4; ++dt) {
      const int dcol = h * DK_ + dt * 16 + cc;
      O[(size_t)(b * S_ + qg) * D_ + dcol] = (_Float16)(accO[dt][e] * inv);
    }
  }
}

// ---------------------------------------------------------------------------
// Host launcher
// ---------------------------------------------------------------------------
extern "C" void kernel_launch(void* const* d_in, const int* in_sizes, int n_in,
                              void* d_out, int out_size, void* d_ws, size_t ws_size,
                              hipStream_t stream) {
  const float* x  = (const float*)d_in[0];
  const int*   m  = (const int*)  d_in[1];
  const float* Wq = (const float*)d_in[2];
  const float* bq = (const float*)d_in[3];
  const float* Wk = (const float*)d_in[4];
  const float* bk = (const float*)d_in[5];
  const float* Wv = (const float*)d_in[6];
  const float* bv = (const float*)d_in[7];
  const float* Wo = (const float*)d_in[8];
  const float* bo = (const float*)d_in[9];
  float* out = (float*)d_out;

  const int M = B_ * S_;           // 8192
  const size_t xBytes = (size_t)M * D_ * sizeof(_Float16);    // 16 MB
  const size_t wBytes = (size_t)D_ * D_ * sizeof(_Float16);   // 2 MB
  const size_t qBytes = (size_t)B_ * H_ * S_ * DK_ * sizeof(_Float16); // 16 MB

  char* ws = (char*)d_ws;
  _Float16* x16 = (_Float16*)ws;            ws += xBytes;   // reused as O16
  _Float16* WqT = (_Float16*)ws;            ws += wBytes;
  _Float16* WkT = (_Float16*)ws;            ws += wBytes;
  _Float16* WvT = (_Float16*)ws;            ws += wBytes;
  _Float16* WoT = (_Float16*)ws;            ws += wBytes;
  _Float16* Q16 = (_Float16*)ws;            ws += qBytes;
  _Float16* K16 = (_Float16*)ws;            ws += qBytes;
  _Float16* V16 = (_Float16*)ws;            ws += qBytes;
  _Float16* O16 = x16;  // x16 dead after QKV GEMMs; alias for attention output

  // 1) Convert x to f16
  {
    const int n4 = (M * D_) / 4;
    cvt_f16_kernel<<<dim3((n4 + 255) / 256), dim3(256), 0, stream>>>(x, x16, n4);
  }
  // 1b) Transpose-convert weights
  {
    dim3 g(D_ / 32, D_ / 32), blk(256);
    cvt_wT_kernel<<<g, blk, 0, stream>>>(Wq, WqT);
    cvt_wT_kernel<<<g, blk, 0, stream>>>(Wk, WkT);
    cvt_wT_kernel<<<g, blk, 0, stream>>>(Wv, WvT);
    cvt_wT_kernel<<<g, blk, 0, stream>>>(Wo, WoT);
  }
  // 2) QKV projections -> [B,H,S,DK] f16
  {
    dim3 g(M / 128, D_ / 128), blk(256);
    gemm_f16_kernel<<<g, blk, 0, stream>>>(x16, WqT, bq, Q16, nullptr, M, D_, D_, 0);
    gemm_f16_kernel<<<g, blk, 0, stream>>>(x16, WkT, bk, K16, nullptr, M, D_, D_, 1);
    gemm_f16_kernel<<<g, blk, 0, stream>>>(x16, WvT, bv, V16, nullptr, M, D_, D_, 2);
  }
  // 3) Flash attention -> O16 [B,S,D] f16
  {
    dim3 g(S_ / 128, B_ * H_), blk(256);
    flash_attn_kernel<<<g, blk, 0, stream>>>(Q16, K16, V16, m, O16,
                                             0.125f /* 1/sqrt(DK) */);
  }
  // 4) Output projection -> f32 d_out
  {
    dim3 g(M / 128, D_ / 128), blk(256);
    gemm_f16_kernel<<<g, blk, 0, stream>>>(O16, WoT, bo, nullptr, out, M, D_, D_, 3);
  }
}
